// MoE_82592221102585
// MI455X (gfx1250) — matrix operand order repaired and checked by reading, not compiled, not run on hardware
//
#include <hip/hip_runtime.h>
#include <stdint.h>

typedef __attribute__((ext_vector_type(16))) __bf16 v16bf;
typedef __attribute__((ext_vector_type(8)))  float  v8f;
typedef __attribute__((ext_vector_type(4)))  int    v4i;

#define DEVFN static __device__ __forceinline__

// ---- problem dims (fixed by reference) ----
#define MOE_N   8192      // B*T tokens
#define MOE_D   1024
#define MOE_H   2048      // 2*D
#define MOE_E   8
#define MOE_NC  2048      // token chunk per gemm pass
#define LEAKY   0.01f

// ---- gemm tiling ----
#define BM 128
#define BN 128
#define BK 32
#define LDS_ROW 80                  // 64B of K-data + 16B pad (bank-conflict-free)
#define MAT_B   (128 * LDS_ROW)     // 10240 B per staged matrix tile
#define STAGE_B (4 * MAT_B)         // Ahi,Alo,Bhi,Blo
#define LDS_ALLOC (2 * STAGE_B)     // double buffered = 81920 B

// ------------------------------------------------------------------
// CDNA5 async global->LDS copy (ASYNCcnt-tracked), builtin or asm
// ------------------------------------------------------------------
DEVFN void async_cp16(void* lds, const void* g) {
#if __has_builtin(__builtin_amdgcn_global_load_async_to_lds_b128)
  // probe-revealed signature: (AS1 int4*, AS3 int4*, i32 offset, i32 cpol)
  __builtin_amdgcn_global_load_async_to_lds_b128(
      (__attribute__((address_space(1))) v4i*)g,
      (__attribute__((address_space(3))) v4i*)lds, 0, 0);
#else
  // LDS generic address truncates to the LDS byte offset (ISA 10.2 aperture rule)
  unsigned l32 = (unsigned)(uintptr_t)lds;
  asm volatile("global_load_async_to_lds_b128 %0, %1, off"
               :: "v"(l32), "v"(g)
               : "memory");
#endif
}

template <int N> DEVFN void wait_asynccnt() {
#if __has_builtin(__builtin_amdgcn_s_wait_asynccnt)
  __builtin_amdgcn_s_wait_asynccnt(N);
#else
  asm volatile("s_wait_asynccnt %0" :: "i"(N) : "memory");
#endif
}

DEVFN v8f wmma_bf16(v16bf a, v16bf b, v8f c) {
  return __builtin_amdgcn_wmma_f32_16x16x32_bf16(false, a, false, b, (short)0, c,
                                                 false, false);
}

// A fragment (16x32 bf16): lanes 0-15 take K0..7 / K16..23, lanes 16-31 K8..15 / K24..31
DEVFN v16bf frag_a(const char* rowbase, int lane) {
  v16bf r;
  const char* p = rowbase + (lane & 15) * LDS_ROW + ((lane >> 4) << 4);
  ((uint4*)&r)[0] = *(const uint4*)(p);
  ((uint4*)&r)[1] = *(const uint4*)(p + 32);
  return r;
}
// B fragment (32x16 bf16), B staged N-major: lane n<16 -> K0..7 / K16..23 of col n, lane n+16 -> K8..15 / K24..31
DEVFN v16bf frag_b(const char* rowbase, int lane) {
  v16bf r;
  const char* p = rowbase + (lane & 15) * LDS_ROW + ((lane >> 4) << 4);
  ((uint4*)&r)[0] = *(const uint4*)(p);
  ((uint4*)&r)[1] = *(const uint4*)(p + 32);
  return r;
}

// ------------------------------------------------------------------
// GEMM: C[M,Nout] = leaky(A[M,K] @ B_T[Nout,K]^T + bias)
//  A,B given as bf16 hi/lo splits (fp32-faithful: hihi + hilo + lohi)
//  MODE 0: write C as bf16 hi/lo (h activations)
//  MODE 1: out[row_offset+m, n] += gate[row_offset+m, expert] * C
// ------------------------------------------------------------------
template <int MODE>
__global__ __launch_bounds__(256) void moe_gemm(
    const __bf16* __restrict__ Ahi, const __bf16* __restrict__ Alo,
    const __bf16* __restrict__ Bhi, const __bf16* __restrict__ Blo,
    const float* __restrict__ bias, int K, int Nout,
    __bf16* __restrict__ Ohi, __bf16* __restrict__ Olo,
    float* __restrict__ outp, const float* __restrict__ gate,
    int expert, int row_offset)
{
  extern __shared__ char smem[];
  const int tid  = threadIdx.x;
  const int lane = tid & 31;
  const int w    = tid >> 5;
  const int wm   = w & 1;       // 2 waves along M (64 rows each)
  const int wn   = w >> 1;      // 4 waves along N (32 cols each)
  const int m0   = blockIdx.y * BM;
  const int n0   = blockIdx.x * BN;

  const __bf16* srcs[4] = {
      Ahi + (size_t)m0 * K, Alo + (size_t)m0 * K,
      Bhi + (size_t)n0 * K, Blo + (size_t)n0 * K };

  v8f c[4][2];
#pragma unroll
  for (int mt = 0; mt < 4; ++mt)
#pragma unroll
    for (int nt = 0; nt < 2; ++nt)
      c[mt][nt] = (v8f){0.f, 0.f, 0.f, 0.f, 0.f, 0.f, 0.f, 0.f};

  // 2048 16B-chunks per stage (4 matrices x 128 rows x 4), 8 per thread
  auto issue = [&](int stage, int k0) {
    char* sb = smem + stage * STAGE_B;
#pragma unroll
    for (int i = 0; i < 8; ++i) {
      const int ch  = tid + (i << 8);
      const int m   = ch >> 9;
      const int idx = ch & 511;
      const int row = idx >> 2;
      const int cc  = idx & 3;
      const __bf16* g = srcs[m] + (size_t)row * K + (k0 + cc * 8);
      char* l = sb + m * MAT_B + row * LDS_ROW + cc * 16;
      async_cp16(l, g);
    }
  };

  auto compute = [&](int stage) {
    const char* sb  = smem + stage * STAGE_B;
    const char* sAh = sb;
    const char* sAl = sb + MAT_B;
    const char* sBh = sb + 2 * MAT_B;
    const char* sBl = sb + 3 * MAT_B;
    v16bf ah[4], al[4], bh[2], bl[2];
#pragma unroll
    for (int mt = 0; mt < 4; ++mt) {
      const int ro = (wm * 64 + mt * 16) * LDS_ROW;
      ah[mt] = frag_a(sAh + ro, lane);
      al[mt] = frag_a(sAl + ro, lane);
    }
#pragma unroll
    for (int nt = 0; nt < 2; ++nt) {
      const int ro = (wn * 32 + nt * 16) * LDS_ROW;
      bh[nt] = frag_b(sBh + ro, lane);
      bl[nt] = frag_b(sBl + ro, lane);
    }
#pragma unroll
    for (int mt = 0; mt < 4; ++mt)
#pragma unroll
      for (int nt = 0; nt < 2; ++nt) {
        v8f acc = c[mt][nt];
        acc = wmma_bf16(ah[mt], bh[nt], acc);   // hi*hi
        acc = wmma_bf16(ah[mt], bl[nt], acc);   // hi*lo
        acc = wmma_bf16(al[mt], bh[nt], acc);   // lo*hi
        c[mt][nt] = acc;
      }
  };

  const int KT = K >> 5;
  issue(0, 0);
  for (int kt = 0; kt < KT; ++kt) {
    if (kt + 1 < KT) {
      issue((kt + 1) & 1, (kt + 1) * BK);
      wait_asynccnt<8>();   // keep only next stage in flight -> current stage landed
    } else {
      wait_asynccnt<0>();
    }
    __syncthreads();
    compute(kt & 1);
    __syncthreads();        // buffer free for reuse
  }

  // ---- epilogue ----
#pragma unroll
  for (int mt = 0; mt < 4; ++mt) {
#pragma unroll
    for (int nt = 0; nt < 2; ++nt) {
      const int rbase = m0 + wm * 64 + mt * 16 + ((lane >> 4) << 3);
      const int col   = n0 + wn * 32 + nt * 16 + (lane & 15);
      const float bv  = bias[col];
      v8f acc = c[mt][nt];
#pragma unroll
      for (int r = 0; r < 8; ++r) {
        float v = acc[r] + bv;
        v = v > 0.0f ? v : LEAKY * v;
        const int row = rbase + r;
        if (MODE == 0) {
          const __bf16 hi = (__bf16)v;
          const __bf16 lo = (__bf16)(v - (float)hi);
          Ohi[(size_t)row * Nout + col] = hi;
          Olo[(size_t)row * Nout + col] = lo;
        } else {
          const float gv = gate[(size_t)(row_offset + row) * MOE_E + expert];
          float* o = outp + (size_t)(row_offset + row) * Nout + col;
          *o += gv * v;
        }
      }
    }
  }
}

// ------------------------------------------------------------------
// Router: logits -> softmax -> top2 -> sparse dense-gate + prob sums
// ------------------------------------------------------------------
__global__ __launch_bounds__(256) void moe_router(
    const float* __restrict__ x, const float* __restrict__ rw,
    const float* __restrict__ rb, float* __restrict__ gate,
    float* __restrict__ gsum)
{
  const int lane = threadIdx.x & 31;
  const int n = blockIdx.x * 8 + (threadIdx.x >> 5);
  float acc[MOE_E];
#pragma unroll
  for (int e = 0; e < MOE_E; ++e) acc[e] = 0.f;
  const float* xr = x + (size_t)n * MOE_D;
  for (int d = lane; d < MOE_D; d += 32) {
    const float xv = xr[d];
    const float* wr = rw + (size_t)d * MOE_E;
#pragma unroll
    for (int e = 0; e < MOE_E; ++e) acc[e] += xv * wr[e];
  }
#pragma unroll
  for (int off = 16; off > 0; off >>= 1)
#pragma unroll
    for (int e = 0; e < MOE_E; ++e) acc[e] += __shfl_xor(acc[e], off, 32);
  if (lane == 0) {
    float p[MOE_E];
    float mx = -3.4e38f;
#pragma unroll
    for (int e = 0; e < MOE_E; ++e) { p[e] = acc[e] + rb[e]; mx = fmaxf(mx, p[e]); }
    float s = 0.f;
#pragma unroll
    for (int e = 0; e < MOE_E; ++e) { p[e] = expf(p[e] - mx); s += p[e]; }
    const float inv = 1.f / s;
#pragma unroll
    for (int e = 0; e < MOE_E; ++e) p[e] *= inv;
    int i1 = 0;
#pragma unroll
    for (int e = 1; e < MOE_E; ++e) if (p[e] > p[i1]) i1 = e;
    int i2 = (i1 == 0) ? 1 : 0;
#pragma unroll
    for (int e = 0; e < MOE_E; ++e) if (e != i1 && p[e] > p[i2]) i2 = e;
    float* gr = gate + (size_t)n * MOE_E;
#pragma unroll
    for (int e = 0; e < MOE_E; ++e) {
      gr[e] = (e == i1 || e == i2) ? p[e] : 0.f;
      atomicAdd(&gsum[e], p[e]);
    }
  }
}

__global__ void moe_loss(const float* __restrict__ gsum, float* __restrict__ lossp) {
  float s = 0.f;
  const float invN = 1.0f / (float)MOE_N;
#pragma unroll
  for (int e = 0; e < MOE_E; ++e) {
    const float d = (1.0f / (float)MOE_E) - gsum[e] * invN;
    s += d * d;
  }
  *lossp = (s / (float)MOE_E) * 1e-4f;
}

// fp32 -> bf16 hi/lo split, elementwise
__global__ void moe_split(const float* __restrict__ src, __bf16* __restrict__ hi,
                          __bf16* __restrict__ lo, size_t n) {
  for (size_t i = blockIdx.x * (size_t)blockDim.x + threadIdx.x; i < n;
       i += (size_t)gridDim.x * blockDim.x) {
    const float v = src[i];
    const __bf16 h = (__bf16)v;
    hi[i] = h;
    lo[i] = (__bf16)(v - (float)h);
  }
}

// fp32 src[Kdim][Ndim] -> bf16 hi/lo dst[Ndim][Kdim] (transpose + split)
__global__ void moe_split_t(const float* __restrict__ src, __bf16* __restrict__ hi,
                            __bf16* __restrict__ lo, int Kdim, int Ndim) {
  const size_t total = (size_t)Kdim * Ndim;
  for (size_t i = blockIdx.x * (size_t)blockDim.x + threadIdx.x; i < total;
       i += (size_t)gridDim.x * blockDim.x) {
    const int n = (int)(i / Kdim);
    const int k = (int)(i % Kdim);
    const float v = src[(size_t)k * Ndim + n];
    const __bf16 h = (__bf16)v;
    hi[i] = h;                 // i == n*Kdim + k
    lo[i] = (__bf16)(v - (float)h);
  }
}

// ------------------------------------------------------------------
extern "C" void kernel_launch(void* const* d_in, const int* in_sizes, int n_in,
                              void* d_out, int out_size, void* d_ws, size_t ws_size,
                              hipStream_t stream) {
  (void)in_sizes; (void)n_in; (void)out_size; (void)ws_size;
  const float* x  = (const float*)d_in[0];
  const float* rw = (const float*)d_in[1];
  const float* rb = (const float*)d_in[2];
  const float* w1 = (const float*)d_in[3];
  const float* b1 = (const float*)d_in[4];
  const float* w2 = (const float*)d_in[5];
  const float* b2 = (const float*)d_in[6];
  float* out = (float*)d_out;

  char* ws = (char*)d_ws;
  size_t off = 0;
  auto alloc = [&](size_t bytes) -> char* {
    char* p = ws + off;
    off = (off + bytes + 255) & ~(size_t)255;
    return p;
  };
  float*  gsum = (float*)alloc(256);
  float*  gate = (float*)alloc((size_t)MOE_N * MOE_E * sizeof(float));
  __bf16* xhi  = (__bf16*)alloc((size_t)MOE_N * MOE_D * 2);
  __bf16* xlo  = (__bf16*)alloc((size_t)MOE_N * MOE_D * 2);
  __bf16* w1hi = (__bf16*)alloc((size_t)MOE_H * MOE_D * 2);
  __bf16* w1lo = (__bf16*)alloc((size_t)MOE_H * MOE_D * 2);
  __bf16* w2hi = (__bf16*)alloc((size_t)MOE_D * MOE_H * 2);
  __bf16* w2lo = (__bf16*)alloc((size_t)MOE_D * MOE_H * 2);
  __bf16* hhi  = (__bf16*)alloc((size_t)MOE_NC * MOE_H * 2);
  __bf16* hlo  = (__bf16*)alloc((size_t)MOE_NC * MOE_H * 2);

  (void)hipMemsetAsync(d_out, 0, ((size_t)MOE_N * MOE_D + 1) * sizeof(float), stream);
  (void)hipMemsetAsync(gsum, 0, 256, stream);

  moe_router<<<MOE_N / 8, 256, 0, stream>>>(x, rw, rb, gate, gsum);
  moe_split<<<4096, 256, 0, stream>>>(x, xhi, xlo, (size_t)MOE_N * MOE_D);

  (void)hipFuncSetAttribute(reinterpret_cast<const void*>(moe_gemm<0>),
                            hipFuncAttributeMaxDynamicSharedMemorySize, LDS_ALLOC);
  (void)hipFuncSetAttribute(reinterpret_cast<const void*>(moe_gemm<1>),
                            hipFuncAttributeMaxDynamicSharedMemorySize, LDS_ALLOC);

  for (int e = 0; e < MOE_E; ++e) {
    // weights -> transposed bf16 hi/lo slabs (K-contiguous rows)
    moe_split_t<<<2048, 256, 0, stream>>>(w1 + (size_t)e * MOE_D * MOE_H,
                                          w1hi, w1lo, MOE_D, MOE_H);
    moe_split_t<<<2048, 256, 0, stream>>>(w2 + (size_t)e * MOE_H * MOE_D,
                                          w2hi, w2lo, MOE_H, MOE_D);
    for (int ch = 0; ch < MOE_N / MOE_NC; ++ch) {
      const int r0 = ch * MOE_NC;
      moe_gemm<0><<<dim3(MOE_H / BN, MOE_NC / BM), 256, LDS_ALLOC, stream>>>(
          xhi + (size_t)r0 * MOE_D, xlo + (size_t)r0 * MOE_D,
          w1hi, w1lo, b1 + (size_t)e * MOE_H, MOE_D, MOE_H,
          hhi, hlo, nullptr, nullptr, 0, 0);
      moe_gemm<1><<<dim3(MOE_D / BN, MOE_NC / BM), 256, LDS_ALLOC, stream>>>(
          hhi, hlo, w2hi, w2lo, b2 + (size_t)e * MOE_D, MOE_H, MOE_D,
          nullptr, nullptr, out, gate, e, r0);
    }
  }
  moe_loss<<<1, 1, 0, stream>>>(gsum, out + (size_t)MOE_N * MOE_D);
}
